// ChannelSSM_89661737271891
// MI455X (gfx1250) — matrix operand order repaired and checked by reading, not compiled, not run on hardware
//
#include <hip/hip_runtime.h>
#include <hip/hip_bf16.h>

typedef __attribute__((ext_vector_type(16))) _Float16 v16h;
typedef __attribute__((ext_vector_type(8)))  _Float16 v8h;
typedef __attribute__((ext_vector_type(2)))  _Float16 h2;
typedef __attribute__((ext_vector_type(8)))  float    v8f;
typedef __attribute__((ext_vector_type(4)))  unsigned int u32x4;
typedef __attribute__((ext_vector_type(8)))  int      i32x8;
typedef __attribute__((ext_vector_type(4)))  int      i32x4;

union F16x16 { v16h v; v8h q[2]; _Float16 h[16]; };
union F32x8  { v8f  v; float f[8]; };

#define B_   4
#define C_   192
#define H_   64
#define W_   64
#define L_   4096
#define DI_  384
#define N_   16
#define R_   12
#define K_   4
#define EPS_ 1e-5f
#define LDS_STRIDE 40   // 32 f16 data + 4-DWORD pad -> 80B rows, 16B aligned, conflict-free b128

#if defined(__has_builtin)
#if __has_builtin(__builtin_amdgcn_tensor_load_to_lds) && __has_builtin(__builtin_amdgcn_s_wait_tensorcnt)
#define USE_TDM 1
#endif
#endif
#ifndef USE_TDM
#define USE_TDM 0
#endif

__device__ __forceinline__ float sigmoidf_(float v){ return 1.f/(1.f+__expf(-v)); }
__device__ __forceinline__ float siluf_(float v){ return v*sigmoidf_(v); }
__device__ __forceinline__ float softplusf_(float v){ return v>20.f ? v : log1pf(__expf(v)); }

// Direction map: scan step t -> canonical row-major spatial index (H=W=64)
__device__ __forceinline__ int scan_map(int k, int t){
  switch (k) {
    case 0:  return t;
    case 1:  return ((t & 63) << 6) | (t >> 6);
    case 2:  return L_ - 1 - t;
    default: { int u = L_ - 1 - t; return ((u & 63) << 6) | (u >> 6); }
  }
}

// A-fragment from padded row-major LDS tile: two b128 loads per lane.
__device__ __forceinline__ void load_afrag(const _Float16* rowp, int half, F16x16& af){
  const v8h* p = (const v8h*)rowp;       // rowp 16B aligned (stride 80B)
  af.q[0] = p[half];                     // K = 8*half + 0..7
  af.q[1] = p[2 + half];                 // K = 16 + 8*half + 0..7
}
// B-fragment from f16 weights, row-major (N,Kdim): two b128 loads per lane.
__device__ __forceinline__ void load_bfrag(const _Float16* wrow, int k0, int half, F16x16& bf){
  const v8h* p = (const v8h*)(wrow + k0);
  bf.q[0] = p[half];
  bf.q[1] = p[2 + half];
}

// ---------------- f32 -> f16 weight conversion ----------------
__global__ __launch_bounds__(256) void cvt_f16(const float* __restrict__ s, _Float16* __restrict__ d, int n){
  int i = blockIdx.x * 256 + threadIdx.x;
  if (i < n) d[i] = (_Float16)s[i];
}

// ---------------- pooled mean + y1 (SE-like branch) ----------------
__global__ __launch_bounds__(256) void pool_kernel(const float* __restrict__ x, float* __restrict__ pooled){
  __shared__ float s[256];
  const int bc = blockIdx.x;
  float a = 0.f;
  for (int i = threadIdx.x; i < L_; i += 256) a += x[(size_t)bc*L_ + i];
  s[threadIdx.x] = a; __syncthreads();
  for (int off = 128; off > 0; off >>= 1){
    if (threadIdx.x < off) s[threadIdx.x] += s[threadIdx.x + off];
    __syncthreads();
  }
  if (threadIdx.x == 0) pooled[bc] = s[0] * (1.f / L_);
}

__global__ void y1_kernel(const float* __restrict__ pooled, const float* __restrict__ cbr,
                          const float* __restrict__ gm, const float* __restrict__ bt,
                          const float* __restrict__ mean, const float* __restrict__ var,
                          float* __restrict__ y1){
  int i = blockIdx.x * 256 + threadIdx.x;
  if (i >= B_ * C_) return;
  int b = i / C_, o = i % C_;
  float a = 0.f;
  for (int c = 0; c < C_; ++c) a += pooled[b*C_ + c] * cbr[(o*C_ + c)*9 + 4]; // [o][c][1][1]
  a = (a - mean[o]) * rsqrtf(var[o] + EPS_) * gm[o] + bt[o];
  y1[i] = a > 0.f ? a : 0.f;
}

// ---------------- in_proj GEMM: xz[m,e] = sum_c x[b,c,l]*W[e,c] ----------------
__global__ __launch_bounds__(256) void inproj_gemm(const float* __restrict__ x,
                                                   const _Float16* __restrict__ w16, // (768,192) f16
                                                   float* __restrict__ xin,          // (B,DI,L)
                                                   float* __restrict__ z)            // (B,L,DI)
{
  __shared__ __align__(16) _Float16 lds_a[128][LDS_STRIDE];
  const int tid = threadIdx.x, lane = tid & 31, wave = tid >> 5;
  const int half = lane >> 4;
  const int mblk  = blockIdx.x;
  const int b     = (mblk * 128) >> 12;
  const int lbase = (mblk * 128) & (L_ - 1);
  const int nb    = blockIdx.y * 16 + (lane & 15);
  const _Float16* wrow = w16 + (size_t)nb * C_;

  F32x8 acc;
  #pragma unroll
  for (int i = 0; i < 8; ++i) acc.f[i] = 0.f;

  for (int ks = 0; ks < 6; ++ks){
    const int k0 = ks * 32;
    #pragma unroll
    for (int i = 0; i < 16; ++i){
      int idx = tid + i*256;
      int m = idx & 127, kk = idx >> 7;     // coalesced along m; transposed LDS write
      lds_a[m][kk] = (_Float16) x[((size_t)b*C_ + (k0 + kk))*L_ + lbase + m];
    }
    if (ks < 5) __builtin_prefetch(&x[((size_t)b*C_ + (k0 + 32))*L_ + lbase], 0, 0);
    __syncthreads();
    F16x16 af, bf;
    load_afrag(&lds_a[(wave << 4) + (lane & 15)][0], half, af);
    load_bfrag(wrow, k0, half, bf);
    acc.v = __builtin_amdgcn_wmma_f32_16x16x32_f16(false, af.v, false, bf.v,
                                                   (short)0, acc.v, false, false);
    __syncthreads();
  }
  const int rbase = (wave << 4) + (half << 3);
  #pragma unroll
  for (int v = 0; v < 8; ++v){
    int l = lbase + rbase + v;
    float val = acc.f[v];
    if (nb < DI_) xin[((size_t)b*DI_ + nb)*L_ + l] = val;
    else          z[((size_t)b*L_ + l)*DI_ + (nb - DI_)] = val;
  }
}

// ---------------- depthwise 3x3 conv + bias + silu ----------------
__global__ __launch_bounds__(256) void dwconv_silu(const float* __restrict__ xin,
                                                   const float* __restrict__ dww,
                                                   const float* __restrict__ dwb,
                                                   float* __restrict__ xc){
  int idx = blockIdx.x * 256 + threadIdx.x;          // over B*DI*L
  int l  = idx & (L_ - 1);
  int dg = idx >> 12;                                 // b*DI + d
  int d  = dg % DI_;
  int h = l >> 6, w = l & 63;
  float acc = dwb[d];
  #pragma unroll
  for (int i = 0; i < 3; ++i){
    int hh = h + i - 1;
    if (hh < 0 || hh >= H_) continue;
    #pragma unroll
    for (int j = 0; j < 3; ++j){
      int ww = w + j - 1;
      if (ww < 0 || ww >= W_) continue;
      acc += xin[(size_t)dg*L_ + (hh << 6) + ww] * dww[d*9 + i*3 + j];
    }
  }
  xc[idx] = siluf_(acc);
}

// ---------------- x_dbl GEMM: xdbl[bk,c,t] = sum_d xs[b,k,d,t]*xpw[k,c,d] ----------------
__global__ __launch_bounds__(256) void xdbl_gemm(const float* __restrict__ xc,
                                                 const _Float16* __restrict__ xpw16, // (K,44,384) f16
                                                 float* __restrict__ xdbl)           // (B*K,44,L)
{
  __shared__ __align__(16) _Float16 lds_a[128][LDS_STRIDE];
  const int tid = threadIdx.x, lane = tid & 31, wave = tid >> 5;
  const int half = lane >> 4;
  const int bk = blockIdx.z, b = bk >> 2, k = bk & 3;
  const int l0 = blockIdx.x * 128;
  const int c  = blockIdx.y * 16 + (lane & 15);
  const int cc = c < 44 ? c : 43;                    // clamp OOB reads
  const _Float16* wrow = xpw16 + ((size_t)k*44 + cc) * DI_;

  F32x8 acc;
  #pragma unroll
  for (int i = 0; i < 8; ++i) acc.f[i] = 0.f;

  for (int ks = 0; ks < 12; ++ks){
    const int k0 = ks * 32;
    #pragma unroll
    for (int i = 0; i < 16; ++i){
      int idx = tid + i*256;
      int m = idx & 127, kk = idx >> 7;
      int lc = scan_map(k, l0 + m);
      lds_a[m][kk] = (_Float16) xc[((size_t)b*DI_ + (k0 + kk))*L_ + lc];
    }
    __syncthreads();
    F16x16 af, bf;
    load_afrag(&lds_a[(wave << 4) + (lane & 15)][0], half, af);
    load_bfrag(wrow, k0, half, bf);
    acc.v = __builtin_amdgcn_wmma_f32_16x16x32_f16(false, af.v, false, bf.v,
                                                   (short)0, acc.v, false, false);
    __syncthreads();
  }
  if (c < 44){
    const int rbase = (wave << 4) + (half << 3);
    #pragma unroll
    for (int v = 0; v < 8; ++v)
      xdbl[((size_t)bk*44 + c)*L_ + l0 + rbase + v] = acc.f[v];
  }
}

// ---------------- selective scan (one (b,k) per block, one d per thread) ----------------
__global__ __launch_bounds__(384) void scan_kernel(const float* __restrict__ xc,
                                                   const float* __restrict__ xdbl,
                                                   const float* __restrict__ dtpw, // (K,DI,R)
                                                   const float* __restrict__ dtpb, // (K,DI)
                                                   const float* __restrict__ Alog, // (K,DI,N)
                                                   const float* __restrict__ Ds,   // (K,DI)
                                                   float* __restrict__ ycanon)     // (B,K,L,DI)
{
  __shared__ float sh[44];
  const int d  = threadIdx.x;
  const int bk = blockIdx.x, b = bk >> 2, k = bk & 3;

  float Ad[N_], hst[N_], dtw[R_];
  #pragma unroll
  for (int n = 0; n < N_; ++n){
    Ad[n]  = -__expf(Alog[((size_t)k*DI_ + d)*N_ + n]);
    hst[n] = 0.f;
  }
  #pragma unroll
  for (int r = 0; r < R_; ++r) dtw[r] = dtpw[((size_t)k*DI_ + d)*R_ + r];
  const float dtb = dtpb[k*DI_ + d];
  const float Dv  = Ds[k*DI_ + d];

  const float* xrow  = xc + ((size_t)b*DI_ + d) * L_;
  const float* dbase = xdbl + (size_t)bk * 44 * L_;
  float* ybase = ycanon + (size_t)bk * L_ * DI_ + d;

  for (int t = 0; t < L_; ++t){
    __syncthreads();
    if (d < 44) sh[d] = dbase[(size_t)d*L_ + t];
    __syncthreads();
    const int lc = scan_map(k, t);
    const float xt = xrow[lc];
    float a = dtb;
    #pragma unroll
    for (int r = 0; r < R_; ++r) a += dtw[r] * sh[r];
    const float dt  = softplusf_(a);
    const float dtx = dt * xt;
    float y = 0.f;
    #pragma unroll
    for (int n = 0; n < N_; ++n){
      hst[n] = hst[n] * __expf(dt * Ad[n]) + dtx * sh[12 + n];
      y += hst[n] * sh[28 + n];
    }
    ybase[(size_t)lc * DI_] = y + Dv * xt;
  }
}

// ---------------- combine 4 directions + LayerNorm + silu(z) gate -> f16 ----------------
__global__ __launch_bounds__(384) void combine_ln_gate(const float* __restrict__ ycanon,
                                                       const float* __restrict__ z,
                                                       const float* __restrict__ g,
                                                       const float* __restrict__ beta,
                                                       _Float16* __restrict__ yg16) // (B*L,DI) f16
{
  __shared__ float s1[384], s2[384], smv[2];
  const int d  = threadIdx.x;
  const int bl = blockIdx.x;               // b*L + l
  const int b  = bl >> 12;
  const size_t base = (size_t)b*K_*L_*DI_ + (size_t)(bl & (L_ - 1))*DI_ + d;
  const size_t kstep = (size_t)L_ * DI_;
  float v = ycanon[base] + ycanon[base + kstep] + ycanon[base + 2*kstep] + ycanon[base + 3*kstep];
  s1[d] = v; s2[d] = v * v; __syncthreads();
  if (d < 128){ s1[d] += s1[d+128] + s1[d+256]; s2[d] += s2[d+128] + s2[d+256]; }
  __syncthreads();
  for (int off = 64; off > 0; off >>= 1){
    if (d < off){ s1[d] += s1[d+off]; s2[d] += s2[d+off]; }
    __syncthreads();
  }
  if (d == 0){ float mu = s1[0] * (1.f/DI_); smv[0] = mu; smv[1] = s2[0] * (1.f/DI_) - mu*mu; }
  __syncthreads();
  const float mu = smv[0], var = smv[1];
  const float yn = (v - mu) * rsqrtf(var + EPS_) * g[d] + beta[d];
  const float zv = z[(size_t)bl*DI_ + d];
  yg16[(size_t)bl*DI_ + d] = (_Float16)(yn * siluf_(zv));
}

// ---------------- out_proj GEMM: y2[m,o] = sum_d yg[m,d]*W[o,d] ----------------
// A staged via Tensor Data Mover (async DMA, padded LDS rows) when available.
__global__ __launch_bounds__(256) void outproj_gemm(const _Float16* __restrict__ yg16, // (B*L,DI) f16
                                                    const _Float16* __restrict__ w16,  // (192,384) f16
                                                    float* __restrict__ y2)            // (B*L,C)
{
  __shared__ __align__(16) _Float16 lds_a[128][LDS_STRIDE];
  const int tid = threadIdx.x, lane = tid & 31, wave = tid >> 5;
  const int half = lane >> 4;
  const int m0 = blockIdx.x * 128;
  const int o  = blockIdx.y * 16 + (lane & 15);
  const _Float16* wrow = w16 + (size_t)o * DI_;

  F32x8 acc;
  #pragma unroll
  for (int i = 0; i < 8; ++i) acc.f[i] = 0.f;

  for (int ks = 0; ks < 12; ++ks){
    const int k0 = ks * 32;
#if USE_TDM
    if (wave == 0){
      // D# group0: count=1 | lds_addr | 57-bit global tile addr | type=2
      unsigned long long ga = (unsigned long long)(uintptr_t)(yg16 + (size_t)m0*DI_ + k0);
      u32x4 g0;
      g0[0] = 1u;                                        // count = 1 valid descriptor
      g0[1] = (unsigned)(uintptr_t)&lds_a[0][0];         // LDS byte offset (addr[31:0])
      g0[2] = (unsigned)(ga & 0xFFFFFFFFu);              // global_addr[31:0]
      g0[3] = (unsigned)((ga >> 32) & 0x1FFFFFFu) | (2u << 30); // global_addr[56:32], type=2
      // D# group1: data_size=2B, pad every 16 DWORDs by 4 DWORDs (64B row + 16B pad = 80B)
      i32x8 g1;
      g1[0] = (int)((1u << 16) | (1u << 20) | (3u << 22) | (3u << 25));
      g1[1] = (int)(384u << 16);        // tensor_dim0[15:0] = 384 (row length, elems)
      g1[2] = (int)(16384u << 16);      // tensor_dim0 hi=0 | tensor_dim1[15:0] = 16384 rows
      g1[3] = (int)(32u << 16);         // tensor_dim1 hi=0 | tile_dim0 = 32
      g1[4] = 128;                      // tile_dim1 = 128 rows (tile_dim2 = 0)
      g1[5] = 384;                      // tensor_dim0_stride[31:0] = 384 elems
      g1[6] = 0;
      g1[7] = 0;
      i32x4 zz4 = {0, 0, 0, 0};
      i32x8 zz8 = {0, 0, 0, 0, 0, 0, 0, 0};
      __builtin_amdgcn_tensor_load_to_lds(g0, g1, zz4, zz4, zz8, 0);
    }
    __builtin_amdgcn_s_wait_tensorcnt(0);
#else
    #pragma unroll
    for (int i = 0; i < 16; ++i){
      int idx = tid + i*256;
      int m = idx >> 5, kk = idx & 31;
      lds_a[m][kk] = yg16[(size_t)(m0 + m)*DI_ + k0 + kk];
    }
#endif
    __syncthreads();
    F16x16 af, bf;
    load_afrag(&lds_a[(wave << 4) + (lane & 15)][0], half, af);
    load_bfrag(wrow, k0, half, bf);
    acc.v = __builtin_amdgcn_wmma_f32_16x16x32_f16(false, af.v, false, bf.v,
                                                   (short)0, acc.v, false, false);
    __syncthreads();
  }
  const int rbase = (wave << 4) + (half << 3);
  #pragma unroll
  for (int v = 0; v < 8; ++v)
    y2[(size_t)(m0 + rbase + v)*C_ + o] = acc.f[v];
}

// ---------------- final: out = x * (y1 + y2) ----------------
__global__ __launch_bounds__(256) void final_kernel(const float* __restrict__ x,
                                                    const float* __restrict__ y1,
                                                    const float* __restrict__ y2,
                                                    float* __restrict__ out){
  int idx = blockIdx.x * 256 + threadIdx.x;  // over B*C*L
  int l  = idx & (L_ - 1);
  int bc = idx >> 12;                        // b*C + c
  int b = bc / C_, c = bc % C_;
  out[idx] = x[idx] * (y1[bc] + y2[((size_t)b*L_ + l)*C_ + c]);
}

extern "C" void kernel_launch(void* const* d_in, const int* in_sizes, int n_in,
                              void* d_out, int out_size, void* d_ws, size_t ws_size,
                              hipStream_t stream) {
  (void)in_sizes; (void)n_in; (void)out_size; (void)ws_size;
  const float* x     = (const float*)d_in[0];
  const float* cbr_w = (const float*)d_in[1];
  const float* bn_g  = (const float*)d_in[2];
  const float* bn_b  = (const float*)d_in[3];
  const float* bn_m  = (const float*)d_in[4];
  const float* bn_v  = (const float*)d_in[5];
  const float* ipw   = (const float*)d_in[6];
  const float* dww   = (const float*)d_in[7];
  const float* dwb   = (const float*)d_in[8];
  const float* xpw   = (const float*)d_in[9];
  const float* dtpw  = (const float*)d_in[10];
  const float* dtpb  = (const float*)d_in[11];
  const float* Alog  = (const float*)d_in[12];
  const float* Ds    = (const float*)d_in[13];
  const float* lng   = (const float*)d_in[14];
  const float* lnb   = (const float*)d_in[15];
  const float* opw   = (const float*)d_in[16];
  float* out = (float*)d_out;

  float* ws     = (float*)d_ws;
  float* pooled = ws;                                   // 768
  float* y1     = pooled + 768;                         // 768
  float* xin    = y1 + 768;                             // B*DI*L
  float* zbuf   = xin    + (size_t)B_*DI_*L_;           // B*L*DI
  float* xcbuf  = zbuf   + (size_t)B_*L_*DI_;           // B*DI*L
  float* xdbl   = xcbuf  + (size_t)B_*DI_*L_;           // B*K*44*L
  float* ycanon = xdbl   + (size_t)B_*K_*44*L_;         // B*K*L*DI
  float* y2     = ycanon + (size_t)B_*K_*L_*DI_;        // B*L*C
  _Float16* yg16  = (_Float16*)(y2 + (size_t)B_*L_*C_); // B*L*DI halves
  _Float16* ipw16 = yg16  + (size_t)B_*L_*DI_;          // 768*192
  _Float16* xpw16 = ipw16 + (size_t)768*192;            // 4*44*384
  _Float16* opw16 = xpw16 + (size_t)K_*44*DI_;          // 192*384

  cvt_f16<<<dim3((768*192 + 255)/256), 256, 0, stream>>>(ipw, ipw16, 768*192);
  cvt_f16<<<dim3((K_*44*DI_ + 255)/256), 256, 0, stream>>>(xpw, xpw16, K_*44*DI_);
  cvt_f16<<<dim3((C_*DI_ + 255)/256), 256, 0, stream>>>(opw, opw16, C_*DI_);

  pool_kernel<<<dim3(B_*C_), 256, 0, stream>>>(x, pooled);
  y1_kernel<<<dim3(3), 256, 0, stream>>>(pooled, cbr_w, bn_g, bn_b, bn_m, bn_v, y1);
  inproj_gemm<<<dim3(128, 48), 256, 0, stream>>>(x, ipw16, xin, zbuf);
  dwconv_silu<<<dim3((B_*DI_*L_)/256), 256, 0, stream>>>(xin, dww, dwb, xcbuf);
  xdbl_gemm<<<dim3(32, 3, B_*K_), 256, 0, stream>>>(xcbuf, xpw16, xdbl);
  scan_kernel<<<dim3(B_*K_), 384, 0, stream>>>(xcbuf, xdbl, dtpw, dtpb, Alog, Ds, ycanon);
  combine_ln_gate<<<dim3(B_*L_), 384, 0, stream>>>(ycanon, zbuf, lng, lnb, yg16);
  outproj_gemm<<<dim3(128, 12), 256, 0, stream>>>(yg16, opw16, y2);
  final_kernel<<<dim3((B_*C_*L_)/256), 256, 0, stream>>>(x, y1, y2, out);
}
